// MultiHeadAttention_63273458205427
// MI455X (gfx1250) — matrix operand order repaired and checked by reading, not compiled, not run on hardware
//
#include <hip/hip_runtime.h>
#include <hip/hip_bf16.h>

typedef _Float16 f16;
typedef __attribute__((ext_vector_type(16))) _Float16 v16h;
typedef __attribute__((ext_vector_type(8)))  float    v8f;

#define B_  4
#define S_  2048
#define H_  16
#define HD_ 64
#define V_  1024

// ---------------- WMMA helpers (gfx1250 wave32 layouts) ----------------

__device__ __forceinline__ v8f wmma(v16h a, v16h b, v8f c) {
  return __builtin_amdgcn_wmma_f32_16x16x32_f16(false, a, false, b, (short)0, c,
                                                false, false);
}

// A-matrix 16x32 f16 from row-major f16 [16 x ld] tile.
// lane: m = l&15, kgroup = (l>>4)*16; K = kg+0..15.
__device__ __forceinline__ v16h ldA_h(const f16* __restrict__ tile, int ld) {
  int l = threadIdx.x & 31;
  int m = l & 15, kg = (l >> 4) << 4;
  const f16* r = tile + (size_t)m * ld;
  v16h a;
#pragma unroll
  for (int i = 0; i < 16; ++i) { a[i] = r[kg + i]; }
  return a;
}

// Same but source is fp32 row-major, converted to f16 (only used on x).
__device__ __forceinline__ v16h ldA_f(const float* __restrict__ tile, int ld) {
  int l = threadIdx.x & 31;
  int m = l & 15, kg = (l >> 4) << 4;
  const float* r = tile + (size_t)m * ld;
  v16h a;
#pragma unroll
  for (int i = 0; i < 16; ++i) {
    a[i]     = (f16)r[kg + i];
  }
  return a;
}

// B-matrix 32x16 f16: memory such that element (K=k, N=n) lives at
// p + n*nstride + k  (k contiguous per lane). lane: n = l&15, K = (l>>4)*16 + i.
__device__ __forceinline__ v16h ldB_h(const f16* __restrict__ p, int nstride) {
  int l = threadIdx.x & 31;
  const f16* q = p + (size_t)(l & 15) * nstride + ((l >> 4) << 4);
  v16h b;
#pragma unroll
  for (int i = 0; i < 16; ++i) b[i] = q[i];
  return b;
}

// ---------------- Kernel 0: one-time fp32 -> f16 weight conversion ----
__global__ __launch_bounds__(256) void cvt_w_kernel(
    const float* __restrict__ wqkv, const float* __restrict__ wout,
    f16* __restrict__ WQ, f16* __restrict__ WO) {
  int i = blockIdx.x * blockDim.x + threadIdx.x;
  float4 v = ((const float4*)wout)[i];
  f16* d = WO + (size_t)i * 4;
  d[0] = (f16)v.x; d[1] = (f16)v.y; d[2] = (f16)v.z; d[3] = (f16)v.w;
  if (i < (3 * HD_ * HD_) / 4) {
    float4 q = ((const float4*)wqkv)[i];
    f16* dq = WQ + (size_t)i * 4;
    dq[0] = (f16)q.x; dq[1] = (f16)q.y; dq[2] = (f16)q.z; dq[3] = (f16)q.w;
  }
}

// ---------------- Kernel 1: QKV projection ----------------
// y[b,s,h,e] = sum_d x[b,s,h*64+d] * w_qkv[e,d]   (e in [0,192))
// Q (scaled by 1/8) -> f16 [B,H,S,64];  K -> f16 [B,H,S,64];  V -> f16 [B,H,64,S]
__global__ __launch_bounds__(256) void qkv_kernel(
    const float* __restrict__ x, const f16* __restrict__ w,
    f16* __restrict__ Q, f16* __restrict__ K, f16* __restrict__ VT) {
  int wid = blockIdx.x * (blockDim.x >> 5) + (threadIdx.x >> 5);
  int st = wid & (S_ / 16 - 1);
  int h  = (wid >> 7) & (H_ - 1);
  int b  = wid >> 11;
  int s0 = st << 4;
  int l = threadIdx.x & 31;
  int n = l & 15;
  int row0 = (l >> 4) << 3;

  const float* xt = x + (size_t)(b * S_ + s0) * V_ + h * HD_;
  v16h a0 = ldA_f(xt, V_);        // d = 0..31
  v16h a1 = ldA_f(xt + 32, V_);   // d = 32..63

  size_t bh = (size_t)(b * H_ + h);
#pragma unroll
  for (int et = 0; et < 12; ++et) {
    v8f c = {};
    c = wmma(a0, ldB_h(w + et * 16 * HD_, HD_), c);
    c = wmma(a1, ldB_h(w + et * 16 * HD_ + 32, HD_), c);
    int e = et * 16 + n;
    if (e < 64) {                  // Q with 1/sqrt(64) folded in
      f16* dst = Q + (bh * S_ + s0 + row0) * HD_ + e;
#pragma unroll
      for (int r = 0; r < 8; ++r) dst[(size_t)r * HD_] = (f16)(c[r] * 0.125f);
    } else if (e < 128) {          // K
      f16* dst = K + (bh * S_ + s0 + row0) * HD_ + (e - 64);
#pragma unroll
      for (int r = 0; r < 8; ++r) dst[(size_t)r * HD_] = (f16)c[r];
    } else {                       // V transposed: VT[d][s]
      f16* dst = VT + (bh * HD_ + (e - 128)) * S_ + s0 + row0;
#pragma unroll
      for (int r = 0; r < 8; ++r) dst[r] = (f16)c[r];
    }
  }
}

// ---------------- Kernel 2: causal flash attention ----------------
// One wave handles one (b,h) x 16 query rows; loops over 32-key blocks.
// P tile is double-buffered in LDS so stores(i+1) never alias loads(i):
// only a store->load s_wait_dscnt fence is needed, and the compiler is free
// to overlap next-block global loads with ctx WMMAs.
#define PS 40  // LDS P-tile row stride in halves (80B, keeps 16B chunk alignment)
__global__ __launch_bounds__(256) void attn_kernel(
    const f16* __restrict__ Q, const f16* __restrict__ K,
    const f16* __restrict__ VT, f16* __restrict__ CTX) {
  __shared__ __align__(16) f16 Pt[8][2 * 16 * PS];
  int wv  = threadIdx.x >> 5;
  int wid = blockIdx.x * (blockDim.x >> 5) + wv;
  int st = wid & (S_ / 16 - 1);
  int h  = (wid >> 7) & (H_ - 1);
  int b  = wid >> 11;
  int q0 = st << 4;
  int l = threadIdx.x & 31;
  int n = l & 15;
  int row0 = (l >> 4) << 3;

  size_t bh = (size_t)(b * H_ + h);
  const f16* Qb = Q + (bh * S_ + q0) * HD_;
  const f16* Kb = K + bh * S_ * HD_;
  const f16* Vb = VT + bh * HD_ * S_;

  v16h qa0 = ldA_h(Qb, HD_);
  v16h qa1 = ldA_h(Qb + 32, HD_);

  v8f O0 = {}, O1 = {}, O2 = {}, O3 = {};
  float m[8], lsum[8];
#pragma unroll
  for (int r = 0; r < 8; ++r) { m[r] = -1e30f; lsum[r] = 0.0f; }

  int nkb = (q0 + 47) >> 5;  // key blocks of 32, causal bound
  for (int kb = 0; kb < nkb; ++kb) {
    int k0 = kb << 5;
    f16* P = &Pt[wv][(kb & 1) * 16 * PS];
    // prefetch next key block (K rows: 32 lanes x 128B; VT rows: 64B chunks)
    if (kb + 1 < nkb) {
      __builtin_prefetch(Kb + (size_t)(k0 + 32 + l) * HD_, 0, 0);
      __builtin_prefetch(Vb + (size_t)(2 * l) * S_ + k0 + 32, 0, 0);
    }
    // scores: two 16-key subtiles, WMMA-K = head dim (two 32-slices)
    v8f s0t = {}, s1t = {};
    s0t = wmma(qa0, ldB_h(Kb + (size_t)k0 * HD_, HD_), s0t);
    s0t = wmma(qa1, ldB_h(Kb + (size_t)k0 * HD_ + 32, HD_), s0t);
    s1t = wmma(qa0, ldB_h(Kb + (size_t)(k0 + 16) * HD_, HD_), s1t);
    s1t = wmma(qa1, ldB_h(Kb + (size_t)(k0 + 16) * HD_ + 32, HD_), s1t);

    // issue V B-operand loads early; latency hides under softmax + LDS trip
    v16h vb0 = ldB_h(Vb + k0, S_);
    v16h vb1 = ldB_h(Vb + 16 * S_ + k0, S_);
    v16h vb2 = ldB_h(Vb + 32 * S_ + k0, S_);
    v16h vb3 = ldB_h(Vb + 48 * S_ + k0, S_);

    // causal mask + online softmax row stats
#pragma unroll
    for (int r = 0; r < 8; ++r) {
      int qi = q0 + row0 + r;
      float a  = (k0 + n      <= qi) ? s0t[r] : -1e30f;
      float bv = (k0 + 16 + n <= qi) ? s1t[r] : -1e30f;
      float t = fmaxf(a, bv);
#pragma unroll
      for (int mk = 8; mk >= 1; mk >>= 1) t = fmaxf(t, __shfl_xor(t, mk, 32));
      float mnew = fmaxf(m[r], t);
      float sc = __expf(m[r] - mnew);
      m[r] = mnew;
      float p0 = __expf(a - mnew);
      float p1 = __expf(bv - mnew);
      s0t[r] = p0; s1t[r] = p1;
      float rs = p0 + p1;
#pragma unroll
      for (int mk = 8; mk >= 1; mk >>= 1) rs += __shfl_xor(rs, mk, 32);
      lsum[r] = lsum[r] * sc + rs;
      O0[r] *= sc; O1[r] *= sc; O2[r] *= sc; O3[r] *= sc;
    }

    // C-layout P -> LDS (row-major 16x32), then reload in A-layout
#pragma unroll
    for (int r = 0; r < 8; ++r) {
      P[(row0 + r) * PS + n]      = (f16)s0t[r];
      P[(row0 + r) * PS + 16 + n] = (f16)s1t[r];
    }
    asm volatile("s_wait_dscnt 0" ::: "memory");
    v16h pa = ldA_h(P, PS);
    // ctx accumulate: WMMA-K = 32 keys, four 16-wide d tiles from VT
    O0 = wmma(pa, vb0, O0);
    O1 = wmma(pa, vb1, O1);
    O2 = wmma(pa, vb2, O2);
    O3 = wmma(pa, vb3, O3);
  }

  // normalize and write ctx f16 [B,S,1024] row-major
#pragma unroll
  for (int r = 0; r < 8; ++r) {
    float inv = 1.0f / lsum[r];
    f16* d = CTX + (size_t)(b * S_ + q0 + row0 + r) * V_ + h * HD_;
    d[n]      = (f16)(O0[r] * inv);
    d[16 + n] = (f16)(O1[r] * inv);
    d[32 + n] = (f16)(O2[r] * inv);
    d[48 + n] = (f16)(O3[r] * inv);
  }
}

// ---------------- Kernel 3: output projection ----------------
// out[row, o] = sum_v ctx[row, v] * w_out16[o, v];  wave = (s-tile, 64-output block)
__global__ __launch_bounds__(256) void proj_kernel(
    const f16* __restrict__ CTX, const f16* __restrict__ WO,
    float* __restrict__ out) {
  int wid = blockIdx.x * (blockDim.x >> 5) + (threadIdx.x >> 5);
  int eb    = wid & 15;    // 64-wide output block
  int stile = wid >> 4;    // 16 rows of flattened [B*S]
  int l = threadIdx.x & 31;
  int n = l & 15;
  int row0 = (l >> 4) << 3;

  const f16* Ab = CTX + (size_t)stile * 16 * V_;
  v8f c0 = {}, c1 = {}, c2 = {}, c3 = {};
#pragma unroll 2
  for (int ks = 0; ks < V_ / 32; ++ks) {
    v16h a = ldA_h(Ab + ks * 32, V_);
    const f16* wb = WO + (size_t)eb * 64 * V_ + ks * 32;
    c0 = wmma(a, ldB_h(wb, V_), c0);
    c1 = wmma(a, ldB_h(wb + 16 * V_, V_), c1);
    c2 = wmma(a, ldB_h(wb + 32 * V_, V_), c2);
    c3 = wmma(a, ldB_h(wb + 48 * V_, V_), c3);
  }
#pragma unroll
  for (int r = 0; r < 8; ++r) {
    float* ob = out + (size_t)(stile * 16 + row0 + r) * V_ + eb * 64;
    ob[n]      = c0[r];
    ob[16 + n] = c1[r];
    ob[32 + n] = c2[r];
    ob[48 + n] = c3[r];
  }
}

// ---------------- host side ----------------
extern "C" void kernel_launch(void* const* d_in, const int* in_sizes, int n_in,
                              void* d_out, int out_size, void* d_ws, size_t ws_size,
                              hipStream_t stream) {
  const float* x     = (const float*)d_in[0];
  const float* w_qkv = (const float*)d_in[1];
  const float* w_out = (const float*)d_in[2];
  (void)in_sizes; (void)n_in; (void)out_size; (void)ws_size;

  const size_t nQ = (size_t)B_ * H_ * S_ * HD_;  // 8M f16 elements each
  f16* Q    = (f16*)d_ws;
  f16* K    = Q + nQ;
  f16* VT   = K + nQ;
  f16* CTX  = VT + nQ;                 // [B*S, 1024] f16
  f16* WO16 = CTX + nQ;                // 1M f16 (2 MB)
  f16* WQ16 = WO16 + (size_t)V_ * V_;  // 12288 f16

  cvt_w_kernel<<<(V_ * V_ / 4) / 256, 256, 0, stream>>>(w_qkv, w_out, WQ16, WO16);

  const int waves = B_ * H_ * (S_ / 16);  // 8192
  const int blocks = waves / 8;           // 256 threads = 8 waves per block
  qkv_kernel <<<blocks, 256, 0, stream>>>(x, WQ16, Q, K, VT);
  attn_kernel<<<blocks, 256, 0, stream>>>(Q, K, VT, CTX);
  proj_kernel<<<blocks, 256, 0, stream>>>(CTX, WO16, (float*)d_out);
}
